// ESNTest_15668040695915
// MI455X (gfx1250) — matrix operand order repaired and checked by reading, not compiled, not run on hardware
//
#include <hip/hip_runtime.h>
#include <hip/hip_bf16.h>
#include <math.h>

// ---------------------------------------------------------------------------
// ESN forward pass for MI455X (gfx1250, wave32, WMMA).
//   h_t = tanh(Win @ x_t + Wh @ h_{t-1});  out[t-100] = [h_t, x_t]
// Sequential in t -> 10000 GEMV steps. Wh (64MB fp32-equivalent) is L2
// resident (192MB L2), so each step is L2-bandwidth bound. We use the
// bf16 split trick (W = W_hi + W_lo, h = h_hi + h_lo, 3 WMMAs) to get
// ~fp32 accuracy on the v_wmma_f32_16x16x32_bf16 pipe with the same
// byte traffic as an fp32 GEMV.
// ---------------------------------------------------------------------------

#define R_DIM   4096
#define I_DIM   8
#define T_STEPS 10000
#define WARMUP  100
#define OUT_W   (R_DIM + I_DIM)

typedef __attribute__((ext_vector_type(16))) __bf16 v16bf;
typedef __attribute__((ext_vector_type(8)))  __bf16 v8bf;
typedef __attribute__((ext_vector_type(8)))  float  v8f;

// ---------------------------------------------------------------------------
// Split fp32 Wh into bf16 hi/lo planes: w ~= hi + lo with ~16 mantissa bits.
// ---------------------------------------------------------------------------
__global__ __launch_bounds__(256) void esn_split_wh(
    const float* __restrict__ Wh,
    __bf16* __restrict__ Whi, __bf16* __restrict__ Wlo, int n)
{
    int idx = blockIdx.x * blockDim.x + threadIdx.x;
    if (idx < n) {
        float w  = Wh[idx];
        __bf16 h = (__bf16)w;
        Whi[idx] = h;
        Wlo[idx] = (__bf16)(w - (float)h);
    }
}

__global__ __launch_bounds__(256) void esn_init_h(
    __bf16* __restrict__ hhi, __bf16* __restrict__ hlo)
{
    int idx = blockIdx.x * blockDim.x + threadIdx.x;
    if (idx < R_DIM) {
        hhi[idx] = (__bf16)0.0f;
        hlo[idx] = (__bf16)0.0f;
    }
}

// ---------------------------------------------------------------------------
// One recurrence step. Block = 256 threads = 8 waves; block owns 16 output
// rows [bid*16, bid*16+16). Wave w reduces K-range [w*512, w*512+512) with
// WMMA (A = h broadcast over M; B = Wh tile, row-major contiguous loads).
// Wave partials are summed through LDS; wave 0 applies Win@x_t + tanh and
// writes h (fp32 to out, bf16 hi/lo for the next step).
// ---------------------------------------------------------------------------
__global__ __launch_bounds__(256) void esn_step(
    const __bf16* __restrict__ Whi, const __bf16* __restrict__ Wlo,
    const __bf16* __restrict__ hphi, const __bf16* __restrict__ hplo,
    __bf16* __restrict__ hnhi, __bf16* __restrict__ hnlo,
    const float* __restrict__ Win, const float* __restrict__ x,
    float* __restrict__ out, int t)
{
    __shared__ float sums[8][16];

    const int tid   = threadIdx.x;
    const int wave  = tid >> 5;
    const int lane  = tid & 31;
    const int ncol  = lane & 15;      // B column -> output row within tile
    const int khalf = lane >> 4;      // which 16-wide K half this lane holds
    const int rowBase = blockIdx.x * 16;

    // B fragment source: lane holds Wh[rowBase+ncol][k + khalf*16 .. +15]
    const __bf16* bHiRow = Whi + (size_t)(rowBase + ncol) * R_DIM + khalf * 16;
    const __bf16* bLoRow = Wlo + (size_t)(rowBase + ncol) * R_DIM + khalf * 16;
    // A fragment source offsets within h (contiguous K order, matching B):
    // elems 0..7 = h[k + aoff .. +7], elems 8..15 = h[k + 8 + aoff .. +7]
    const int aoff = khalf * 16;

    const int k0 = wave * 512;
    const int k1 = k0 + 512;

    v8f acc = {};

    for (int k = k0; k < k1; k += 32) {
        v8bf a0h = *(const v8bf*)(hphi + k + aoff);
        v8bf a1h = *(const v8bf*)(hphi + k + 8 + aoff);
        v8bf a0l = *(const v8bf*)(hplo + k + aoff);
        v8bf a1l = *(const v8bf*)(hplo + k + 8 + aoff);
        v16bf Ahi = __builtin_shufflevector(a0h, a1h,
                        0,1,2,3,4,5,6,7,8,9,10,11,12,13,14,15);
        v16bf Alo = __builtin_shufflevector(a0l, a1l,
                        0,1,2,3,4,5,6,7,8,9,10,11,12,13,14,15);
        v16bf Bhi = *(const v16bf*)(bHiRow + k);
        v16bf Blo = *(const v16bf*)(bLoRow + k);

        // W_hi*h_hi + W_lo*h_hi + W_hi*h_lo  (lo*lo term ~2^-16, dropped)
        acc = __builtin_amdgcn_wmma_f32_16x16x32_bf16(
                  false, Ahi, false, Bhi, (short)0, acc, false, false);
        acc = __builtin_amdgcn_wmma_f32_16x16x32_bf16(
                  false, Ahi, false, Blo, (short)0, acc, false, false);
        acc = __builtin_amdgcn_wmma_f32_16x16x32_bf16(
                  false, Alo, false, Bhi, (short)0, acc, false, false);
    }

    // All C rows are identical (A rows broadcast); acc[0] on lane L carries
    // the K-partial of y[rowBase + L%16]. Lane halves are duplicates.
    if (lane < 16) sums[wave][lane] = acc[0];
    __syncthreads();

    if (wave == 0 && lane < 16) {
        float y = 0.0f;
        #pragma unroll
        for (int w = 0; w < 8; ++w) y += sums[w][lane];

        const int r = rowBase + lane;
        const float* xt   = x   + (size_t)t * I_DIM;
        const float* winr = Win + (size_t)r * I_DIM;
        float u = 0.0f;
        #pragma unroll
        for (int i = 0; i < I_DIM; ++i) u += winr[i] * xt[i];

        float h = tanhf(y + u);

        __bf16 hh = (__bf16)h;
        hnhi[r] = hh;
        hnlo[r] = (__bf16)(h - (float)hh);

        if (t >= WARMUP)
            out[(size_t)(t - WARMUP) * OUT_W + r] = h;
    }

    // One block copies x_t into the tail columns of the output row.
    if (blockIdx.x == 0 && wave == 1 && lane < I_DIM && t >= WARMUP)
        out[(size_t)(t - WARMUP) * OUT_W + R_DIM + lane] =
            x[(size_t)t * I_DIM + lane];
}

// ---------------------------------------------------------------------------
// Host launcher. Inputs: x [T,I] f32, Win [R,I] f32, Wh [R,R] f32, warmup i32.
// Workspace: Whi (32MB) + Wlo (32MB) + 2x ping-pong h hi/lo (32KB).
// ---------------------------------------------------------------------------
extern "C" void kernel_launch(void* const* d_in, const int* in_sizes, int n_in,
                              void* d_out, int out_size, void* d_ws, size_t ws_size,
                              hipStream_t stream)
{
    const float* x   = (const float*)d_in[0];
    const float* Win = (const float*)d_in[1];
    const float* Wh  = (const float*)d_in[2];
    float* out = (float*)d_out;

    const size_t nW = (size_t)R_DIM * R_DIM;
    size_t off = 0;
    char* ws = (char*)d_ws;

    __bf16* Whi = (__bf16*)(ws + off); off += nW * sizeof(__bf16);
    __bf16* Wlo = (__bf16*)(ws + off); off += nW * sizeof(__bf16);
    __bf16* hhi[2];
    __bf16* hlo[2];
    for (int b = 0; b < 2; ++b) {
        hhi[b] = (__bf16*)(ws + off); off += R_DIM * sizeof(__bf16);
        hlo[b] = (__bf16*)(ws + off); off += R_DIM * sizeof(__bf16);
    }
    if (ws_size < off) return;  // workspace too small; nothing safe to do

    esn_split_wh<<<(int)((nW + 255) / 256), 256, 0, stream>>>(Wh, Whi, Wlo, (int)nW);
    esn_init_h<<<(R_DIM + 255) / 256, 256, 0, stream>>>(hhi[0], hlo[0]);

    for (int t = 0; t < T_STEPS; ++t) {
        const int cur = t & 1;
        const int nxt = cur ^ 1;
        esn_step<<<R_DIM / 16, 256, 0, stream>>>(
            Whi, Wlo, hhi[cur], hlo[cur], hhi[nxt], hlo[nxt],
            Win, x, out, t);
    }
}
